// GD_Unroll_41120016892381
// MI455X (gfx1250) — matrix operand order repaired and checked by reading, not compiled, not run on hardware
//
#include <hip/hip_runtime.h>
#include <hip/hip_bf16.h>

#define N_NODES 50000
#define N_EDGES 800000
#define DIM     256
#define GD_STEPS 5
#define ROW_TILES 3125   // 50000 / 16 exactly

typedef __attribute__((ext_vector_type(16))) __bf16 v16bf;
typedef __attribute__((ext_vector_type(8)))  float  v8f;

// Pack two 8-float contiguous chunks into a v16bf fragment (RNE packed converts).
__device__ __forceinline__ v16bf frag16(const float* __restrict__ p0,
                                        const float* __restrict__ p1) {
    float4 a = *(const float4*)(p0);
    float4 b = *(const float4*)(p0 + 4);
    float4 c = *(const float4*)(p1);
    float4 d = *(const float4*)(p1 + 4);
    v16bf r;
    r[0]  = (__bf16)a.x; r[1]  = (__bf16)a.y; r[2]  = (__bf16)a.z; r[3]  = (__bf16)a.w;
    r[4]  = (__bf16)b.x; r[5]  = (__bf16)b.y; r[6]  = (__bf16)b.z; r[7]  = (__bf16)b.w;
    r[8]  = (__bf16)c.x; r[9]  = (__bf16)c.y; r[10] = (__bf16)c.z; r[11] = (__bf16)c.w;
    r[12] = (__bf16)d.x; r[13] = (__bf16)d.y; r[14] = (__bf16)d.z; r[15] = (__bf16)d.w;
    return r;
}

// deg[col] += 1  (u32 atomics -> global_atomic_add_u32)
__global__ void deg_kernel(const int* __restrict__ col, unsigned* __restrict__ deg) {
    int i = blockIdx.x * blockDim.x + threadIdx.x;
    if (i < N_EDGES) atomicAdd(&deg[col[i]], 1u);
}

__global__ void dinv_kernel(const unsigned* __restrict__ deg, float* __restrict__ dinv) {
    int i = blockIdx.x * blockDim.x + threadIdx.x;
    if (i < N_NODES) {
        float d = (float)deg[i];
        dinv[i] = (d > 0.0f) ? rsqrtf(d) : 0.0f;
    }
}

// Per-edge norm precompute (loop-invariant across the 5 GD steps).
__global__ void norm_kernel(const int* __restrict__ row, const int* __restrict__ col,
                            const float* __restrict__ dinv, float* __restrict__ nrm) {
    int i = blockIdx.x * blockDim.x + threadIdx.x;
    if (i < N_EDGES) nrm[i] = dinv[row[i]] * dinv[col[i]];
}

// Loop-invariant weight pre-convert: fp32 -> bf16, row-major (all 5 steps at once).
__global__ void wconv_kernel(const float* __restrict__ src, __bf16* __restrict__ dst, int n) {
    int i = blockIdx.x * blockDim.x + threadIdx.x;
    if (i < n) dst[i] = (__bf16)src[i];
}

// agg[col] += nrm[e] * x[row]; one wave per edge, 8 floats/lane.
__global__ void scatter_kernel(const float* __restrict__ X,
                               const int* __restrict__ row,
                               const int* __restrict__ col,
                               const float* __restrict__ nrm,
                               float* __restrict__ agg) {
    int e = blockIdx.x * 8 + (threadIdx.x >> 5);
    if (e >= N_EDGES) return;
    int lane = threadIdx.x & 31;
    int r = row[e];
    int c = col[e];
    float w = nrm[e];
    const float* xs = X + (size_t)r * DIM + lane * 8;
    float* ad = agg + (size_t)c * DIM + lane * 8;
    float4 v0 = *(const float4*)(xs);
    float4 v1 = *(const float4*)(xs + 4);
    unsafeAtomicAdd(ad + 0, w * v0.x);
    unsafeAtomicAdd(ad + 1, w * v0.y);
    unsafeAtomicAdd(ad + 2, w * v0.z);
    unsafeAtomicAdd(ad + 3, w * v0.w);
    unsafeAtomicAdd(ad + 4, w * v1.x);
    unsafeAtomicAdd(ad + 5, w * v1.y);
    unsafeAtomicAdd(ad + 6, w * v1.z);
    unsafeAtomicAdd(ad + 7, w * v1.w);
}

// Out = X @ W0k^T + A @ W1k^T. bf16 WMMA, f32 accumulate. Weights pre-converted
// to bf16 so each B fragment is a single 32-byte load (two b128), no converts.
// Block = 256 threads = 8 waves; wave w: rowTile = blockIdx.x*2 + (w>>2),
// 64-col group = (w&3). Each wave: 16x64 output tile, 4 C-tiles share one A frag.
__global__ void gemm_kernel(const float* __restrict__ X,
                            const float* __restrict__ A,
                            const __bf16* __restrict__ W0k,
                            const __bf16* __restrict__ W1k,
                            float* __restrict__ Out) {
    int w       = threadIdx.x >> 5;
    int lane    = threadIdx.x & 31;
    int rowTile = blockIdx.x * 2 + (w >> 2);
    if (rowTile >= ROW_TILES) return;          // uniform per wave: EXEC stays all-ones
    int r0   = rowTile * 16;
    int n0   = (w & 3) * 64;
    int half = lane >> 4;                       // lane group within wave
    int m    = lane & 15;

    v8f acc[4];
    v8f z = {};
#pragma unroll
    for (int i = 0; i < 4; ++i) acc[i] = z;

    const float*  srcs[2] = { X,   A   };
    const __bf16* wts [2] = { W0k, W1k };

#pragma unroll
    for (int s = 0; s < 2; ++s) {
        const float*  xrow = srcs[s] + (size_t)(r0 + m) * DIM;
        const __bf16* Wk   = wts[s];
#pragma unroll
        for (int kk = 0; kk < DIM; kk += 32) {
            // A 16x32 bf16 frag: half0 -> K {kk..kk+15}; half1 -> K {kk+16..kk+31}
            const float* ap = xrow + kk + half * 16;
            v16bf afrag = frag16(ap, ap + 8);
#pragma unroll
            for (int ct = 0; ct < 4; ++ct) {
                // B 32x16 frag: lane col n, 16 contiguous bf16 K from W row n (32B aligned)
                int n = n0 + ct * 16 + m;
                const __bf16* bp = Wk + (size_t)n * DIM + kk + half * 16;
                v16bf bfrag = *(const v16bf*)bp;
                acc[ct] = __builtin_amdgcn_wmma_f32_16x16x32_bf16(
                    false, afrag, false, bfrag, (short)0, acc[ct], false, false);
            }
        }
    }

    // D layout: VGPR r -> row (r0 + half*8 + r), col = n
#pragma unroll
    for (int ct = 0; ct < 4; ++ct) {
        int n = n0 + ct * 16 + m;
        float* o = Out + (size_t)(r0 + half * 8) * DIM + n;
#pragma unroll
        for (int r = 0; r < 8; ++r) o[(size_t)r * DIM] = acc[ct][r];
    }
}

extern "C" void kernel_launch(void* const* d_in, const int* in_sizes, int n_in,
                              void* d_out, int out_size, void* d_ws, size_t ws_size,
                              hipStream_t stream) {
    (void)in_sizes; (void)n_in; (void)out_size; (void)ws_size;

    const float* x_in = (const float*)d_in[0];
    const int*   ei   = (const int*)d_in[1];     // [2, E] flat: rows then cols
    const float* W0   = (const float*)d_in[2];   // [5,256,256] fp32
    const float* W1   = (const float*)d_in[3];
    float* out = (float*)d_out;

    const size_t XB = (size_t)N_NODES * DIM * sizeof(float);   // 51.2 MB
    const int    WN = GD_STEPS * DIM * DIM;                    // 327680 elems per weight set
    char* ws = (char*)d_ws;
    float*    xA   = (float*)(ws);                             // ping-pong buffer
    float*    agg  = (float*)(ws + XB);                        // aggregation buffer
    unsigned* deg  = (unsigned*)(ws + 2 * XB);
    float*    dinv = (float*)(ws + 2 * XB + (size_t)N_NODES * 4);
    float*    nrm  = (float*)(ws + 2 * XB + (size_t)N_NODES * 8);
    __bf16*   W0bf = (__bf16*)(ws + 2 * XB + (size_t)N_NODES * 8
                               + (size_t)N_EDGES * 4);         // 32B-aligned
    __bf16*   W1bf = W0bf + WN;

    const int* row = ei;
    const int* col = ei + N_EDGES;

    // One-time (per launch) loop-invariant precomputes
    hipMemsetAsync(deg, 0, (size_t)N_NODES * sizeof(unsigned), stream);
    deg_kernel <<<(N_EDGES + 255) / 256, 256, 0, stream>>>(col, deg);
    dinv_kernel<<<(N_NODES + 255) / 256, 256, 0, stream>>>(deg, dinv);
    norm_kernel<<<(N_EDGES + 255) / 256, 256, 0, stream>>>(row, col, dinv, nrm);
    wconv_kernel<<<(WN + 255) / 256, 256, 0, stream>>>(W0, W0bf, WN);
    wconv_kernel<<<(WN + 255) / 256, 256, 0, stream>>>(W1, W1bf, WN);

    // Ping-pong through d_out so step 4 (odd count) lands in d_out:
    // in -> out -> xA -> out -> xA -> out
    const float* cur = x_in;
    for (int k = 0; k < GD_STEPS; ++k) {
        float* nxt = (k % 2 == 0) ? out : xA;
        hipMemsetAsync(agg, 0, XB, stream);
        scatter_kernel<<<N_EDGES / 8, 256, 0, stream>>>(cur, row, col, nrm, agg);
        gemm_kernel<<<(ROW_TILES + 1) / 2, 256, 0, stream>>>(
            cur, agg, W0bf + (size_t)k * DIM * DIM, W1bf + (size_t)k * DIM * DIM, nxt);
        cur = nxt;
    }
}
